// Actor_58248346468980
// MI455X (gfx1250) — matrix operand order repaired and checked by reading, not compiled, not run on hardware
//
#include <hip/hip_runtime.h>
#include <hip/hip_bf16.h>

#define VOCAB 32000
#define EMB   512
#define HID   1024
#define BATCH 64
#define SEQ   64
#define GATES (3 * HID)   // 3072

typedef __attribute__((ext_vector_type(16))) _Float16 v16h;
typedef __attribute__((ext_vector_type(8)))  float    v8f;
typedef __attribute__((ext_vector_type(4)))  unsigned int u32x4;

union Frag { v16h v; u32x4 u[2]; };

__device__ __forceinline__ _Float16 f2h(float f) { return (_Float16)f; }

__device__ __forceinline__ v8f wmma16(const Frag& a, const Frag& b, v8f c) {
  return __builtin_amdgcn_wmma_f32_16x16x32_f16(false, a.v, false, b.v,
                                                (short)0, c, false, false);
}

// ---------------------------------------------------------------------------
// Pack a row-major f32 weight matrix W (N x K) into fp16 WMMA B-fragment
// order. Fragment tile = 16 (N) x 32 (K). Packed layout:
//   Wp[((n_tile*KT + k_tile)*32 + lane)*16 + e]  with
//   K = k_tile*32 + (lane>>4)*8 + (e&7) + (e>>3)*16   (AMD table k order)
//   N = n_tile*16 + (lane&15)   (VGPR v=e>>1 holds two adjacent K)
// so each lane's 16 halfs are CONTIGUOUS -> two b128 loads in the GEMM.
// ---------------------------------------------------------------------------
__global__ __launch_bounds__(256)
void pack_w_kernel(const float* __restrict__ W, _Float16* __restrict__ Wp,
                   int N, int K) {
  unsigned idx = blockIdx.x * 256u + threadIdx.x;
  unsigned total = (unsigned)N * (unsigned)K;
  if (idx >= total) return;
  int KT = K >> 5;
  unsigned tile = idx >> 9;         // /512 elements per tile
  unsigned within = idx & 511u;
  int lane = (int)(within >> 4);
  int e    = (int)(within & 15u);
  int n_tile = (int)(tile / (unsigned)KT);
  int k_tile = (int)(tile % (unsigned)KT);
  int k = k_tile * 32 + (lane >> 4) * 8 + (e & 7) + (e >> 3) * 16;
  int n = n_tile * 16 + (lane & 15);
  Wp[idx] = f2h(W[(size_t)n * K + k]);
}

// ---------------------------------------------------------------------------
// h0 = 0 (f32 + f16), x0 = broadcast emb[0] row (f16)
// ---------------------------------------------------------------------------
__global__ __launch_bounds__(256)
void init_state_kernel(const float* __restrict__ emb, _Float16* __restrict__ x0,
                       float* __restrict__ h0, _Float16* __restrict__ h0h) {
  int i = blockIdx.x * 256 + threadIdx.x;
  if (i < BATCH * HID) { h0[i] = 0.0f; h0h[i] = (_Float16)0.0f; }
  if (i < BATCH * EMB) { x0[i] = f2h(emb[i & (EMB - 1)]); }
}

// ---------------------------------------------------------------------------
// Fused GRU step (gates GEMMs + nonlinearity, no intermediate buffers).
// Each wave owns one (j_tile, m_tile): a 16x16 patch of h_new.
// Only FOUR accumulators are needed: the r and z gates use only the SUM
// (x@Wih + h@Whh), which WMMA accumulates directly across both phases;
// the n gate keeps i_n and h_n separate (n = tanh(i_n + r*h_n)).
// Fragments are double-buffered: kt+1 loads are issued before kt's WMMAs,
// so one full iteration of load latency is hidden even at 1 wave/SIMD.
// Grid: HID/16 j-tiles * 4 m-tiles = 256 waves = 32 blocks of 256 threads.
// h16 in/out and x are double-buffered by the host (cross-wave WAR).
// ---------------------------------------------------------------------------
__global__ __launch_bounds__(256)
void gru_fused_kernel(const _Float16* __restrict__ x,     // 64 x EMB (f16)
                      const _Float16* __restrict__ h16,   // 64 x HID (f16, in)
                      const _Float16* __restrict__ Wih,   // packed GATES x EMB
                      const _Float16* __restrict__ Whh,   // packed GATES x HID
                      const float* __restrict__ b_ih,
                      const float* __restrict__ b_hh,
                      float* __restrict__ h,              // 64 x HID f32 in/out
                      _Float16* __restrict__ h16_out) {   // 64 x HID f16 out
  int lane = (int)(threadIdx.x & 31u);
  int wave = (int)(threadIdx.x >> 5u);
  int id = blockIdx.x * 8 + wave;
  int j_tile = id >> 2;          // 0..63
  int m_tile = id & 3;           // 0..3
  int r  = lane & 15;
  int hi = lane >> 4;

  v8f accR = {}, accZ = {}, accIN = {}, accHN = {};

  const int KTx = EMB >> 5;      // 16
  const int KTh = HID >> 5;      // 32
  const int JT  = HID >> 4;      // 64 j-tiles per gate strip

  // ---- phase 1: accumulate x @ w_ih^T into accR / accZ / accIN ----
  {
    const _Float16* bR = Wih + ((size_t)(0 * JT + j_tile) * KTx) * 512 + (size_t)lane * 16;
    const _Float16* bZ = Wih + ((size_t)(1 * JT + j_tile) * KTx) * 512 + (size_t)lane * 16;
    const _Float16* bN = Wih + ((size_t)(2 * JT + j_tile) * KTx) * 512 + (size_t)lane * 16;
    const _Float16* arow = x + (size_t)(m_tile * 16 + r) * EMB + hi * 8;
    Frag a0, r0, z0, n0;
    a0.u[0] = *(const u32x4*)(arow);
    a0.u[1] = *(const u32x4*)(arow + 16);
    r0.u[0] = *(const u32x4*)(bR); r0.u[1] = *(const u32x4*)(bR + 8);
    z0.u[0] = *(const u32x4*)(bZ); z0.u[1] = *(const u32x4*)(bZ + 8);
    n0.u[0] = *(const u32x4*)(bN); n0.u[1] = *(const u32x4*)(bN + 8);
    for (int kt = 0; kt < KTx - 1; ++kt) {
      Frag a1, r1, z1, n1;
      const _Float16* an = arow + (kt + 1) * 32;
      a1.u[0] = *(const u32x4*)(an);
      a1.u[1] = *(const u32x4*)(an + 16);
      r1.u[0] = *(const u32x4*)(bR + 512); r1.u[1] = *(const u32x4*)(bR + 520); bR += 512;
      z1.u[0] = *(const u32x4*)(bZ + 512); z1.u[1] = *(const u32x4*)(bZ + 520); bZ += 512;
      n1.u[0] = *(const u32x4*)(bN + 512); n1.u[1] = *(const u32x4*)(bN + 520); bN += 512;
      accR  = wmma16(a0, r0, accR);
      accZ  = wmma16(a0, z0, accZ);
      accIN = wmma16(a0, n0, accIN);
      a0 = a1; r0 = r1; z0 = z1; n0 = n1;
    }
    accR  = wmma16(a0, r0, accR);
    accZ  = wmma16(a0, z0, accZ);
    accIN = wmma16(a0, n0, accIN);
  }
  // ---- phase 2: accumulate h @ w_hh^T into accR / accZ (cont.) / accHN ----
  {
    const _Float16* bR = Whh + ((size_t)(0 * JT + j_tile) * KTh) * 512 + (size_t)lane * 16;
    const _Float16* bZ = Whh + ((size_t)(1 * JT + j_tile) * KTh) * 512 + (size_t)lane * 16;
    const _Float16* bN = Whh + ((size_t)(2 * JT + j_tile) * KTh) * 512 + (size_t)lane * 16;
    const _Float16* arow = h16 + (size_t)(m_tile * 16 + r) * HID + hi * 8;
    Frag a0, r0, z0, n0;
    a0.u[0] = *(const u32x4*)(arow);
    a0.u[1] = *(const u32x4*)(arow + 16);
    r0.u[0] = *(const u32x4*)(bR); r0.u[1] = *(const u32x4*)(bR + 8);
    z0.u[0] = *(const u32x4*)(bZ); z0.u[1] = *(const u32x4*)(bZ + 8);
    n0.u[0] = *(const u32x4*)(bN); n0.u[1] = *(const u32x4*)(bN + 8);
    for (int kt = 0; kt < KTh - 1; ++kt) {
      Frag a1, r1, z1, n1;
      const _Float16* an = arow + (kt + 1) * 32;
      a1.u[0] = *(const u32x4*)(an);
      a1.u[1] = *(const u32x4*)(an + 16);
      r1.u[0] = *(const u32x4*)(bR + 512); r1.u[1] = *(const u32x4*)(bR + 520); bR += 512;
      z1.u[0] = *(const u32x4*)(bZ + 512); z1.u[1] = *(const u32x4*)(bZ + 520); bZ += 512;
      n1.u[0] = *(const u32x4*)(bN + 512); n1.u[1] = *(const u32x4*)(bN + 520); bN += 512;
      accR  = wmma16(a0, r0, accR);
      accZ  = wmma16(a0, z0, accZ);
      accHN = wmma16(a0, n0, accHN);
      a0 = a1; r0 = r1; z0 = z1; n0 = n1;
    }
    accR  = wmma16(a0, r0, accR);
    accZ  = wmma16(a0, z0, accZ);
    accHN = wmma16(a0, n0, accHN);
  }

  // ---- epilogue: GRU nonlinearity in registers ----
  // C/D layout: VGPR i -> M = m_tile*16 + hi*8 + i ; N(col) = j_tile*16 + r
  int col = j_tile * 16 + r;
  float bR2 = b_ih[col] + b_hh[col];
  float bZ2 = b_ih[HID + col] + b_hh[HID + col];
  float biN = b_ih[2 * HID + col];
  float bhN = b_hh[2 * HID + col];
#pragma unroll
  for (int i = 0; i < 8; ++i) {
    size_t ofs = (size_t)(m_tile * 16 + hi * 8 + i) * HID + col;
    float rg = 1.0f / (1.0f + __expf(-(accR[i] + bR2)));
    float zg = 1.0f / (1.0f + __expf(-(accZ[i] + bZ2)));
    float ng = tanhf(accIN[i] + biN + rg * (accHN[i] + bhN));
    float hn = (1.0f - zg) * ng + zg * h[ofs];
    h[ofs] = hn;
    h16_out[ofs] = f2h(hn);
  }
}

// ---------------------------------------------------------------------------
// WMMA GEMM:  C(64 x N, f32) = A(64 x K, f16 row-major) * Bp(packed) + bias
// One wave computes a 64x16 strip (4 M-tiles share one B fragment).
// blockDim = 256 (8 waves); grid.x = N / 128.
// ---------------------------------------------------------------------------
__global__ __launch_bounds__(256)
void gemm_wmma_kernel(const _Float16* __restrict__ A,
                      const _Float16* __restrict__ Bp,
                      const float* __restrict__ bias,
                      float* __restrict__ C, int N, int K) {
  int lane = (int)(threadIdx.x & 31u);
  int wave = (int)(threadIdx.x >> 5u);
  int n_tile = blockIdx.x * 8 + wave;
  if (n_tile * 16 >= N) return;          // wave-uniform: EXEC stays all-ones
  int KT = K >> 5;
  int r  = lane & 15;
  int hi = lane >> 4;

  v8f acc0 = {}, acc1 = {}, acc2 = {}, acc3 = {};
  const _Float16* bp = Bp + ((size_t)n_tile * KT) * 512 + (size_t)lane * 16;
  const _Float16* a0 = A + (size_t)(0 * 16 + r) * K;
  const _Float16* a1 = A + (size_t)(1 * 16 + r) * K;
  const _Float16* a2 = A + (size_t)(2 * 16 + r) * K;
  const _Float16* a3 = A + (size_t)(3 * 16 + r) * K;

  for (int kt = 0; kt < KT; ++kt) {
    Frag b;
    b.u[0] = *(const u32x4*)(bp);
    b.u[1] = *(const u32x4*)(bp + 8);
    bp += 512;
    int k0 = kt * 32 + hi * 8;          // A layout: VGPR0-3 K=k0..k0+7, VGPR4-7 +16
    Frag a;
    a.u[0] = *(const u32x4*)(a0 + k0);
    a.u[1] = *(const u32x4*)(a0 + k0 + 16);
    acc0 = wmma16(a, b, acc0);
    a.u[0] = *(const u32x4*)(a1 + k0);
    a.u[1] = *(const u32x4*)(a1 + k0 + 16);
    acc1 = wmma16(a, b, acc1);
    a.u[0] = *(const u32x4*)(a2 + k0);
    a.u[1] = *(const u32x4*)(a2 + k0 + 16);
    acc2 = wmma16(a, b, acc2);
    a.u[0] = *(const u32x4*)(a3 + k0);
    a.u[1] = *(const u32x4*)(a3 + k0 + 16);
    acc3 = wmma16(a, b, acc3);
  }

  // C/D layout: VGPR i -> M = i + hi*8 ; N = n_tile*16 + r
  int col = n_tile * 16 + r;
  float bv = bias ? bias[col] : 0.0f;
#pragma unroll
  for (int i = 0; i < 8; ++i) {
    C[(size_t)(0 * 16 + hi * 8 + i) * N + col] = acc0[i] + bv;
    C[(size_t)(1 * 16 + hi * 8 + i) * N + col] = acc1[i] + bv;
    C[(size_t)(2 * 16 + hi * 8 + i) * N + col] = acc2[i] + bv;
    C[(size_t)(3 * 16 + hi * 8 + i) * N + col] = acc3[i] + bv;
  }
}

// ---------------------------------------------------------------------------
// Per-row online-softmax + Gumbel argmax + outputs + embedding gather.
// Key insight: argmax(behav - log(-log u)) == argmax((draw?0:logit_j) - g_j)
// since -lse / -logV are per-row constants. One block per batch row.
// ---------------------------------------------------------------------------
__global__ __launch_bounds__(256)
void sample_kernel(const float* __restrict__ logits,
                   const float* __restrict__ gumbel_u,
                   const float* __restrict__ eps_u,
                   const float* __restrict__ emb,
                   _Float16* __restrict__ x_next,
                   float* __restrict__ out, int t) {
  int b = blockIdx.x;
  int tid = threadIdx.x;
  const float* lrow = logits + (size_t)b * VOCAB;
  const float* urow = gumbel_u + ((size_t)t * BATCH + b) * VOCAB;
  bool draw = eps_u[t * BATCH + b] <= 0.5f;

  float m = -3.4e38f, s = 0.0f;
  float bestKey = -3.4e38f;
  int bestIdx = 0;
  for (int j = tid; j < VOCAB; j += 256) {
    float x = lrow[j];
    float mo = m;
    m = fmaxf(m, x);
    s = s * __expf(mo - m) + __expf(x - m);
    float u = fminf(fmaxf(urow[j], 1e-12f), 1.0f - 1e-12f);
    float g = __logf(-__logf(u));
    float key = (draw ? 0.0f : x) - g;
    if (key > bestKey) { bestKey = key; bestIdx = j; }
  }

  __shared__ float sm[256], ss[256], sk[256];
  __shared__ int si[256];
  __shared__ int idxSh;
  sm[tid] = m; ss[tid] = s; sk[tid] = bestKey; si[tid] = bestIdx;
  __syncthreads();
  for (int ofs = 128; ofs > 0; ofs >>= 1) {
    if (tid < ofs) {
      float m2 = sm[tid + ofs], s2 = ss[tid + ofs];
      float mn = fmaxf(sm[tid], m2);
      ss[tid] = ss[tid] * __expf(sm[tid] - mn) + s2 * __expf(m2 - mn);
      sm[tid] = mn;
      if (sk[tid + ofs] > sk[tid]) { sk[tid] = sk[tid + ofs]; si[tid] = si[tid + ofs]; }
    }
    __syncthreads();
  }
  if (tid == 0) {
    float lse = sm[0] + __logf(ss[0]);
    int idx = si[0];
    float lp = lrow[idx] - lse;
    float behav = draw ? -__logf((float)VOCAB) : lp;
    float off = fminf(fmaxf(__expf(behav), 0.001f), 1.0f);
    float corr = __expf(lp) / off;
    out[(size_t)b * SEQ + t] = (float)idx;                       // outs.T
    out[BATCH * SEQ + (size_t)b * SEQ + t] = corr;               // corrs.T
    out[2 * BATCH * SEQ + (size_t)b * SEQ + t] = lp;             // lps.T
    idxSh = idx;
  }
  __syncthreads();
  int idx = idxSh;
  const float* erow = emb + (size_t)idx * EMB;
  for (int e = tid; e < EMB; e += 256) x_next[(size_t)b * EMB + e] = f2h(erow[e]);
}

// ---------------------------------------------------------------------------
extern "C" void kernel_launch(void* const* d_in, const int* in_sizes, int n_in,
                              void* d_out, int out_size, void* d_ws, size_t ws_size,
                              hipStream_t stream) {
  const float* emb    = (const float*)d_in[0];
  const float* w_ih   = (const float*)d_in[1];
  const float* w_hh   = (const float*)d_in[2];
  const float* b_ih   = (const float*)d_in[3];
  const float* b_hh   = (const float*)d_in[4];
  const float* w_out  = (const float*)d_in[5];
  const float* b_out  = (const float*)d_in[6];
  const float* gumbel = (const float*)d_in[7];
  const float* eps    = (const float*)d_in[8];
  float* out = (float*)d_out;

  char* ws = (char*)d_ws;
  size_t off = 0;
  auto alloc = [&](size_t bytes) -> void* {
    off = (off + 255) & ~(size_t)255;
    void* p = ws + off;
    off += bytes;
    return p;
  };
  _Float16* w_ih_p  = (_Float16*)alloc((size_t)GATES * EMB * 2);
  _Float16* w_hh_p  = (_Float16*)alloc((size_t)GATES * HID * 2);
  _Float16* w_out_p = (_Float16*)alloc((size_t)VOCAB * HID * 2);
  float*    logits  = (float*)   alloc((size_t)BATCH * VOCAB * 4);
  float*    h_f     = (float*)   alloc((size_t)BATCH * HID * 4);
  _Float16* h_h0    = (_Float16*)alloc((size_t)BATCH * HID * 2);
  _Float16* h_h1    = (_Float16*)alloc((size_t)BATCH * HID * 2);
  _Float16* x_h0    = (_Float16*)alloc((size_t)BATCH * EMB * 2);
  _Float16* x_h1    = (_Float16*)alloc((size_t)BATCH * EMB * 2);
  _Float16* h_buf[2] = { h_h0, h_h1 };
  _Float16* x_buf[2] = { x_h0, x_h1 };

  // --- pack weights to fp16 WMMA fragment layout (L2-resident thereafter) ---
  pack_w_kernel<<<(GATES * EMB + 255) / 256, 256, 0, stream>>>(w_ih, w_ih_p, GATES, EMB);
  pack_w_kernel<<<(GATES * HID + 255) / 256, 256, 0, stream>>>(w_hh, w_hh_p, GATES, HID);
  pack_w_kernel<<<(VOCAB * HID + 255) / 256, 256, 0, stream>>>(w_out, w_out_p, VOCAB, HID);
  init_state_kernel<<<(BATCH * HID + 255) / 256, 256, 0, stream>>>(emb, x_buf[0], h_f, h_buf[0]);

  for (int t = 0; t < SEQ; ++t) {
    const _Float16* xin  = x_buf[t & 1];
    const _Float16* hin  = h_buf[t & 1];
    _Float16*       hout = h_buf[(t + 1) & 1];
    _Float16*       xout = x_buf[(t + 1) & 1];
    // fused gates GEMMs + GRU nonlinearity (256 waves)
    gru_fused_kernel<<<32, 256, 0, stream>>>(xin, hin, w_ih_p, w_hh_p,
                                             b_ih, b_hh, h_f, hout);
    // logits = h_new @ w_out^T + b_out  (dominant GEMM, L2-resident weights)
    gemm_wmma_kernel<<<VOCAB / 128, 256, 0, stream>>>(hout, w_out_p, b_out,
                                                      logits, VOCAB, HID);
    // online softmax + Gumbel argmax + outputs + next-x embedding gather
    sample_kernel<<<BATCH, 256, 0, stream>>>(logits, gumbel, eps, emb, xout, out, t);
  }
  (void)in_sizes; (void)n_in; (void)out_size; (void)ws_size;
}
